// TERMGVPEncoder_80642305950258
// MI455X (gfx1250) — matrix operand order repaired and checked by reading, not compiled, not run on hardware
//
#include <hip/hip_runtime.h>
#include <hip/hip_bf16.h>
#include <math.h>

typedef __attribute__((ext_vector_type(16))) _Float16 v16h;
typedef __attribute__((ext_vector_type(8)))  _Float16 v8h;
typedef __attribute__((ext_vector_type(8)))  float    v8f;

constexpr int   C_B   = 8;
constexpr int   C_N   = 2048;
constexpr int   C_K   = 30;
constexpr int   C_CH  = 148;   // 3*16 + 100
constexpr int   C_NODES = C_B * C_N;          // 16384
constexpr int   C_EDGES = C_B * C_N * C_K;    // 491520
constexpr float C_EPS = 1e-8f;

// ---------------------------------------------------------------------------
// WMMA fragment helpers (CDNA5 wave32 layouts, CDNA5 ISA)
// ---------------------------------------------------------------------------

// 16-bit A fragment 16x32 from row-major f16 [rows x lda] (LDS).
// Per lane: two contiguous 8-half chunks (16B aligned) -> 2x ds_load_b128.
__device__ inline v16h frag_a(const _Float16* A, int lda, int m0, int k0) {
  int lane = threadIdx.x & 31;
  int m   = m0 + (lane & 15);
  int off = (lane & 16) ? 8 : 0;
  const _Float16* row = A + m * lda + k0 + off;
  v8h lo = *(const v8h*)(row);        // K = off..off+7
  v8h hi = *(const v8h*)(row + 16);   // K = 16+off..16+off+7
  return __builtin_shufflevector(lo, hi, 0, 1, 2, 3, 4, 5, 6, 7,
                                         8, 9, 10, 11, 12, 13, 14, 15);
}

// B fragment from pre-packed weights: tile (kt,nt), lane-major, 16 halves
// contiguous per lane (32B aligned) -> wide global loads, zero-pad baked in.
__device__ inline v16h frag_b_packed(const _Float16* B, int NT, int kt, int nt) {
  int lane = threadIdx.x & 31;
  const v16h* p = (const v16h*)(B + (((size_t)(kt * NT + nt) * 32) + lane) * 16);
  return *p;
}

// f32 C/D tile store to LDS [rows x ldc]
__device__ inline void store_c(float* C, int ldc, int m0, int n0, v8f c) {
  int lane = threadIdx.x & 31;
  int col = n0 + (lane & 15);
  int r0  = m0 + ((lane & 16) ? 8 : 0);
#pragma unroll
  for (int r = 0; r < 8; ++r) C[(r0 + r) * ldc + col] = c[r];
}

// C[MT*16 x NT*16] (f32, LDS) = A[MT*16 x KT*32] (f16, LDS) * Bpacked
// Tiles split across the 4 waves of the workgroup; caller handles barriers.
template<int MT, int KT, int NT>
__device__ inline void gemm_tiles(const _Float16* A, int lda,
                                  const _Float16* B, float* C, int ldc) {
  int wave = threadIdx.x >> 5;
  for (int t = wave; t < MT * NT; t += 4) {
    int mt = t / NT, nt = t - mt * NT;
    v8f acc = {};
#pragma unroll
    for (int kt = 0; kt < KT; ++kt) {
      v16h a = frag_a(A, lda, mt * 16, kt * 32);
      v16h b = frag_b_packed(B, NT, kt, nt);
      acc = __builtin_amdgcn_wmma_f32_16x16x32_f16(
          false, a, false, b, (short)0, acc, false, false);
    }
    store_c(C, ldc, mt * 16, nt * 16, acc);
  }
}

// ---------------------------------------------------------------------------
// Weight repack: f32 [K x N] row-major -> f16 B-fragment-packed tiles
// dst[((kt*NT+nt)*32 + lane)*16 + e] = W[kt*32 + (lane&16?8:0) + (e&7) + (e&8?16:0)][nt*16 + (lane&15)]
// ---------------------------------------------------------------------------
__global__ void pack_b_kernel(const float* __restrict__ src,
                              _Float16* __restrict__ dst,
                              int K, int N, int NT, int total) {
  int idx = blockIdx.x * blockDim.x + threadIdx.x;
  if (idx >= total) return;
  int e    = idx & 15;
  int lane = (idx >> 4) & 31;
  int tile = idx >> 9;
  int nt = tile % NT, kt = tile / NT;
  int n = nt * 16 + (lane & 15);
  int k = kt * 32 + ((lane & 16) ? 8 : 0) + (e & 7) + ((e & 8) ? 16 : 0);
  dst[idx] = (k < K && n < N) ? (_Float16)src[k * N + n] : (_Float16)0.f;
}

// ---------------------------------------------------------------------------
// Boundary GVPs (act=False): thread-per-node / thread-per-edge VALU kernels
// ---------------------------------------------------------------------------

template<int SI>
__global__ void node_gvp_kernel(const float* __restrict__ X,
                                const float* __restrict__ wh,
                                const float* __restrict__ wsb,
                                const float* __restrict__ wsw,
                                const float* __restrict__ wv,
                                float* __restrict__ Y, int n) {
  int node = blockIdx.x * blockDim.x + threadIdx.x;
  if (node >= n) return;
  const float* x = X + (size_t)node * (48 + SI);
  float vh[48];
  for (int h = 0; h < 16; ++h)
    for (int c = 0; c < 3; ++c) {
      float a = 0.f;
#pragma unroll
      for (int i = 0; i < 16; ++i) a += x[c * 16 + i] * wh[i * 16 + h];
      vh[c * 16 + h] = a;
    }
  float vn[16];
  for (int h = 0; h < 16; ++h)
    vn[h] = sqrtf(fmaxf(vh[h] * vh[h] + vh[16 + h] * vh[16 + h] + vh[32 + h] * vh[32 + h], C_EPS));
  float* y = Y + (size_t)node * C_CH;
  for (int c = 0; c < 3; ++c)
    for (int o = 0; o < 16; ++o) {
      float a = 0.f;
#pragma unroll
      for (int h = 0; h < 16; ++h) a += vh[c * 16 + h] * wv[h * 16 + o];
      y[c * 16 + o] = a;
    }
  for (int o = 0; o < 100; ++o) {
    float a = wsb[o];
    for (int t = 0; t < SI; ++t) a += x[48 + t] * wsw[t * 100 + o];
#pragma unroll
    for (int h = 0; h < 16; ++h) a += vn[h] * wsw[(SI + h) * 100 + o];
    y[48 + o] = a;
  }
}

__global__ void edge_gvp_kernel(const float* __restrict__ E,
                                const float* __restrict__ wh,
                                const float* __restrict__ wsb,
                                const float* __restrict__ wsw,
                                const float* __restrict__ wv,
                                _Float16* __restrict__ hE, int n) {
  int e = blockIdx.x * blockDim.x + threadIdx.x;
  if (e >= n) return;
  const float* x = E + (size_t)e * C_CH;
  float vh[48];
  for (int h = 0; h < 16; ++h)
    for (int c = 0; c < 3; ++c) {
      float a = 0.f;
#pragma unroll
      for (int i = 0; i < 16; ++i) a += x[c * 16 + i] * wh[i * 16 + h];
      vh[c * 16 + h] = a;
    }
  float vn[16];
  for (int h = 0; h < 16; ++h)
    vn[h] = sqrtf(fmaxf(vh[h] * vh[h] + vh[16 + h] * vh[16 + h] + vh[32 + h] * vh[32 + h], C_EPS));
  _Float16* y = hE + (size_t)e * C_CH;
  for (int c = 0; c < 3; ++c)
    for (int o = 0; o < 16; ++o) {
      float a = 0.f;
#pragma unroll
      for (int h = 0; h < 16; ++h) a += vh[c * 16 + h] * wv[h * 16 + o];
      y[c * 16 + o] = (_Float16)a;
    }
  for (int o = 0; o < 100; ++o) {
    float a = wsb[o];
    for (int t = 0; t < 100; ++t) a += x[48 + t] * wsw[t * 100 + o];
#pragma unroll
    for (int h = 0; h < 16; ++h) a += vn[h] * wsw[(100 + h) * 100 + o];
    y[48 + o] = (_Float16)a;
  }
}

// ---------------------------------------------------------------------------
// Fused per-node message kernel
// ---------------------------------------------------------------------------

struct LayerW {
  const _Float16 *wh0, *ws0, *wv0;   // packed tiles
  const _Float16 *wh1, *ws1, *wv1;
  const _Float16 *wh2, *ws2, *wv2;
  const float *b0, *b1, *b2;         // biases (f32)
  const float *g0, *be0;             // norm0 gamma/beta (f32)
};

constexpr int OFF_AS   = 0;                  // 32x352 f16  scalar A (s + vn1)
constexpr int OFF_AV   = 22528;              // 96x64  f16  vector A (input)
constexpr int OFF_AVH  = 34816;              // 96x64  f16  vh as f16 (A of wv)
constexpr int OFF_CV   = 47104;              // 96x48  f32  vh f32
constexpr int OFF_CS   = 65536;              // 32x112 f32  scalar GEMM out
constexpr int OFF_BS   = 79872;              // 32x128 f16  scalar A (GVP2/3)
constexpr int OFF_AV2  = 88064;              // 96x32  f16  vector A (GVP2/3)
constexpr int OFF_CV2  = 94208;              // 96x16  f32  vout f32
constexpr int OFF_MASK = 100352;             // 32 f32
constexpr int OFF_IDX  = 100480;             // 32 int
constexpr int OFF_DH   = 100608;             // 148 f32
constexpr int OFF_RED  = 101216;             // 4 f32
constexpr int SMEM_MSG = 101504;

__device__ inline void act_pack(const float* Cs, const float* Cv2, const float* bias,
                                _Float16* BsOut, _Float16* Av2Out, int tid) {
  for (int x = tid; x < 32 * 128; x += 128) {
    int e = x >> 7, t = x & 127;
    float v = 0.f;
    if (t < 100) v = fmaxf(Cs[e * 112 + t] + bias[t], 0.f);
    BsOut[x] = (_Float16)v;                       // cols 100..115 = vn, filled later
  }
  for (int x = tid; x < 96 * 32; x += 128) {
    int r = x >> 5, t = x & 31;
    float v = 0.f;
    if (t < 16) {
      int e = r & 31;
      float vx = Cv2[e * 16 + t];
      float vy = Cv2[(32 + e) * 16 + t];
      float vz = Cv2[(64 + e) * 16 + t];
      float nrm = sqrtf(fmaxf(vx * vx + vy * vy + vz * vz, C_EPS));
      float g = 1.f / (1.f + expf(-nrm));
      float mine = (r < 32) ? vx : ((r < 64) ? vy : vz);
      v = mine * g;
    }
    Av2Out[x] = (_Float16)v;
  }
}

__device__ inline void vn_pack(const float* Cv, int ldc, int H,
                               _Float16* vnDst, int vnStride, int vnOff,
                               _Float16* AvhDst, int ldavh, int tid) {
  for (int x = tid; x < 32 * H; x += 128) {
    int e = x / H, h = x - e * H;
    float vx = Cv[e * ldc + h];
    float vy = Cv[(32 + e) * ldc + h];
    float vz = Cv[(64 + e) * ldc + h];
    vnDst[e * vnStride + vnOff + h] =
        (_Float16)sqrtf(fmaxf(vx * vx + vy * vy + vz * vz, C_EPS));
  }
  for (int x = tid; x < 96 * ldavh; x += 128) {
    int r = x / ldavh, t = x - r * ldavh;
    AvhDst[x] = (t < H) ? (_Float16)Cv[r * ldc + t] : (_Float16)0.f;
  }
}

__global__ __launch_bounds__(128)
void msg_layer_kernel(const float* __restrict__ hV_in,
                      const _Float16* __restrict__ hE,
                      const int* __restrict__ E_idx,
                      const float* __restrict__ mask,
                      LayerW W, float* __restrict__ hV_out) {
  extern __shared__ char smem[];
  _Float16* sAs  = (_Float16*)(smem + OFF_AS);
  _Float16* sAv  = (_Float16*)(smem + OFF_AV);
  _Float16* sAvh = (_Float16*)(smem + OFF_AVH);
  float*    sCv  = (float*)  (smem + OFF_CV);
  float*    sCs  = (float*)  (smem + OFF_CS);
  _Float16* sBs  = (_Float16*)(smem + OFF_BS);
  _Float16* sAv2 = (_Float16*)(smem + OFF_AV2);
  float*    sCv2 = (float*)  (smem + OFF_CV2);
  float*    sMask= (float*)  (smem + OFF_MASK);
  int*      sIdx = (int*)    (smem + OFF_IDX);
  float*    sDh  = (float*)  (smem + OFF_DH);
  float*    sRed = (float*)  (smem + OFF_RED);

  const int node = blockIdx.x;
  const int b    = node / C_N;
  const int tid  = threadIdx.x;
  const float mi = mask[node];
  const float* hVc = hV_in + (size_t)node * C_CH;

  // ---- phase 0: pack h_EV (vec [vV|vE|vj], scal [sV|sE|sj]) as f16 A ----
  if (tid < 32) sIdx[tid] = (tid < C_K) ? E_idx[(size_t)node * C_K + tid] : 0;
  for (int x = tid; x < 32 * 352; x += 128) sAs[x] = (_Float16)0.f;
  for (int x = tid; x < 96 * 64;  x += 128) sAv[x] = (_Float16)0.f;
  if (tid < 32) sMask[tid] = 0.f;
  __syncthreads();
  for (int x = tid; x < C_K * 100; x += 128) {
    int e = x / 100, t = x - e * 100;
    int j = sIdx[e];
    const _Float16* he = hE + ((size_t)node * C_K + e) * C_CH;
    sAs[e * 352 + t]       = (_Float16)hVc[48 + t];
    sAs[e * 352 + 100 + t] = he[48 + t];
    sAs[e * 352 + 200 + t] = (_Float16)hV_in[((size_t)b * C_N + j) * C_CH + 48 + t];
  }
  for (int x = tid; x < C_K * 48; x += 128) {
    int e = x / 48, t = x - e * 48;
    int c = t >> 4, i = t & 15;
    int j = sIdx[e];
    const _Float16* he = hE + ((size_t)node * C_K + e) * C_CH;
    int row = (c * 32 + e) * 64;
    sAv[row + i]      = (_Float16)hVc[c * 16 + i];
    sAv[row + 16 + i] = he[c * 16 + i];
    sAv[row + 32 + i] = (_Float16)hV_in[((size_t)b * C_N + j) * C_CH + c * 16 + i];
  }
  if (tid < C_K) sMask[tid] = mi * mask[b * C_N + sIdx[tid]];
  __syncthreads();

  // ---- GVP1 (vi=48,h=48,vo=16,si=300,so=100, act=True) ----
  gemm_tiles<6, 2, 3>(sAv, 64, W.wh0, sCv, 48);          // vh
  __syncthreads();
  vn_pack(sCv, 48, 48, sAs, 352, 300, sAvh, 64, tid);
  __syncthreads();
  gemm_tiles<2, 11, 7>(sAs, 352, W.ws0, sCs, 112);       // scalar out
  gemm_tiles<6, 2, 1>(sAvh, 64, W.wv0, sCv2, 16);        // vout
  __syncthreads();
  act_pack(sCs, sCv2, W.b0, sBs, sAv2, tid);
  __syncthreads();

  // ---- GVP2 (vi=16,h=16,vo=16,si=100,so=100, act=True) ----
  gemm_tiles<6, 1, 1>(sAv2, 32, W.wh1, sCv, 16);
  __syncthreads();
  vn_pack(sCv, 16, 16, sBs, 128, 100, sAvh, 32, tid);
  __syncthreads();
  gemm_tiles<2, 4, 7>(sBs, 128, W.ws1, sCs, 112);
  gemm_tiles<6, 1, 1>(sAvh, 32, W.wv1, sCv2, 16);
  __syncthreads();
  act_pack(sCs, sCv2, W.b1, sBs, sAv2, tid);
  __syncthreads();

  // ---- GVP3 (act=False) ----
  gemm_tiles<6, 1, 1>(sAv2, 32, W.wh2, sCv, 16);
  __syncthreads();
  vn_pack(sCv, 16, 16, sBs, 128, 100, sAvh, 32, tid);
  __syncthreads();
  gemm_tiles<2, 4, 7>(sBs, 128, W.ws2, sCs, 112);
  gemm_tiles<6, 1, 1>(sAvh, 32, W.wv2, sCv2, 16);
  __syncthreads();

  // ---- masked mean-aggregate ----
  for (int o = tid; o < 100; o += 128) {
    float bz = W.b2[o], acc = 0.f;
    for (int e = 0; e < C_K; ++e) acc += (sCs[e * 112 + o] + bz) * sMask[e];
    sDh[48 + o] = acc * (1.0f / C_K);
  }
  for (int t = tid; t < 48; t += 128) {
    int c = t >> 4, i = t & 15;
    float acc = 0.f;
    for (int e = 0; e < C_K; ++e) acc += sCv2[(c * 32 + e) * 16 + i] * sMask[e];
    sDh[t] = acc * (1.0f / C_K);
  }
  __syncthreads();

  // ---- residual + gvp_layernorm(norm0) ----
  if (tid == 0) {
    float msum = 0.f;
    for (int i = 0; i < 16; ++i) {
      float vx = hVc[i] + sDh[i];
      float vy = hVc[16 + i] + sDh[16 + i];
      float vz = hVc[32 + i] + sDh[32 + i];
      msum += vx * vx + vy * vy + vz * vz;
    }
    sRed[0] = 1.f / sqrtf(fmaxf(msum / 16.f, C_EPS));
    float mu = 0.f;
    for (int o = 0; o < 100; ++o) mu += hVc[48 + o] + sDh[48 + o];
    mu *= 0.01f;
    float var = 0.f;
    for (int o = 0; o < 100; ++o) {
      float d = hVc[48 + o] + sDh[48 + o] - mu;
      var += d * d;
    }
    var *= 0.01f;
    sRed[1] = mu;
    sRed[2] = 1.f / sqrtf(var + 1e-5f);
  }
  __syncthreads();
  float* yo = hV_out + (size_t)node * C_CH;
  for (int t = tid; t < 48; t += 128) yo[t] = (hVc[t] + sDh[t]) * sRed[0];
  for (int o = tid; o < 100; o += 128) {
    float v = (hVc[48 + o] + sDh[48 + o] - sRed[1]) * sRed[2];
    yo[48 + o] = W.g0[o] * v + W.be0[o];
  }
}

// ---------------------------------------------------------------------------
// Node FFN (VALU, small): W_dh[0] act, W_dh[1], residual + norm1 + mask
// ---------------------------------------------------------------------------
__global__ __launch_bounds__(128)
void ffn_kernel(const float* __restrict__ hV_in, const float* __restrict__ mask,
                const float* __restrict__ wh0, const float* __restrict__ b0,
                const float* __restrict__ ws0, const float* __restrict__ wv0,
                const float* __restrict__ wh1, const float* __restrict__ b1,
                const float* __restrict__ ws1, const float* __restrict__ wv1,
                const float* __restrict__ gamma, const float* __restrict__ beta,
                float* __restrict__ hV_out) {
  int node = blockIdx.x;
  int tid = threadIdx.x;
  const float* x = hV_in + (size_t)node * C_CH;
  __shared__ float vh[96], vn[32], out1[400], vg[96];
  __shared__ float vh2[96], vn2[32], out2[100], vout2[48], red[4];

  for (int t = tid; t < 96; t += 128) {            // vh = v @ wh0 (16x32)
    int c = t >> 5, h = t & 31;
    float a = 0.f;
#pragma unroll
    for (int i = 0; i < 16; ++i) a += x[c * 16 + i] * wh0[i * 32 + h];
    vh[t] = a;
  }
  __syncthreads();
  for (int h = tid; h < 32; h += 128)
    vn[h] = sqrtf(fmaxf(vh[h] * vh[h] + vh[32 + h] * vh[32 + h] + vh[64 + h] * vh[64 + h], C_EPS));
  __syncthreads();
  for (int o = tid; o < 400; o += 128) {           // ws0 (132 x 400) + relu
    float a = b0[o];
    for (int t = 0; t < 100; ++t) a += x[48 + t] * ws0[t * 400 + o];
    for (int h = 0; h < 32; ++h) a += vn[h] * ws0[(100 + h) * 400 + o];
    out1[o] = fmaxf(a, 0.f);
  }
  for (int t = tid; t < 96; t += 128) {            // vout1 = vh @ wv0 (32x32)
    int c = t >> 5, o = t & 31;
    float a = 0.f;
    for (int h = 0; h < 32; ++h) a += vh[c * 32 + h] * wv0[h * 32 + o];
    vg[t] = a;
  }
  __syncthreads();
  for (int o = tid; o < 32; o += 128) {            // sigmoid-norm gate
    float s2 = vg[o] * vg[o] + vg[32 + o] * vg[32 + o] + vg[64 + o] * vg[64 + o];
    float g = 1.f / (1.f + expf(-sqrtf(fmaxf(s2, C_EPS))));
    vg[o] *= g; vg[32 + o] *= g; vg[64 + o] *= g;
  }
  __syncthreads();
  for (int t = tid; t < 96; t += 128) {            // vh2 = vg @ wh1 (32x32)
    int c = t >> 5, h = t & 31;
    float a = 0.f;
    for (int i = 0; i < 32; ++i) a += vg[c * 32 + i] * wh1[i * 32 + h];
    vh2[t] = a;
  }
  __syncthreads();
  for (int h = tid; h < 32; h += 128)
    vn2[h] = sqrtf(fmaxf(vh2[h] * vh2[h] + vh2[32 + h] * vh2[32 + h] + vh2[64 + h] * vh2[64 + h], C_EPS));
  __syncthreads();
  for (int o = tid; o < 100; o += 128) {           // ws1 (432 x 100)
    float a = b1[o];
    for (int t = 0; t < 400; ++t) a += out1[t] * ws1[t * 100 + o];
    for (int h = 0; h < 32; ++h) a += vn2[h] * ws1[(400 + h) * 100 + o];
    out2[o] = a;
  }
  for (int t = tid; t < 48; t += 128) {            // vout2 = vh2 @ wv1 (32x16)
    int c = t / 16, o = t % 16;
    float a = 0.f;
    for (int h = 0; h < 32; ++h) a += vh2[c * 32 + h] * wv1[h * 16 + o];
    vout2[t] = a;
  }
  __syncthreads();
  if (tid == 0) {
    float msum = 0.f;
    for (int i = 0; i < 16; ++i) {
      float vx = x[i] + vout2[i];
      float vy = x[16 + i] + vout2[16 + i];
      float vz = x[32 + i] + vout2[32 + i];
      msum += vx * vx + vy * vy + vz * vz;
    }
    red[0] = 1.f / sqrtf(fmaxf(msum / 16.f, C_EPS));
    float mu = 0.f;
    for (int o = 0; o < 100; ++o) mu += x[48 + o] + out2[o];
    mu *= 0.01f;
    float var = 0.f;
    for (int o = 0; o < 100; ++o) {
      float d = x[48 + o] + out2[o] - mu;
      var += d * d;
    }
    var *= 0.01f;
    red[1] = mu;
    red[2] = 1.f / sqrtf(var + 1e-5f);
  }
  __syncthreads();
  float mi = mask[node];
  float* y = hV_out + (size_t)node * C_CH;
  for (int t = tid; t < 48; t += 128) y[t] = (x[t] + vout2[t]) * red[0] * mi;
  for (int o = tid; o < 100; o += 128)
    y[48 + o] = (gamma[o] * ((x[48 + o] + out2[o] - red[1]) * red[2]) + beta[o]) * mi;
}

// ---------------------------------------------------------------------------
// Host orchestration. Assumed d_in flatten order (JAX pytree, keys sorted):
//  0:V 1:E 2:E_idx 3:mask
//  4..7  W_e{wh,ws_b,ws_w,wv}  8..11 W_out{...}  12..15 W_v{...}
// per layer l, base = 16 + 24*l:
//  +0..3 W_EV[0]  +4..7 W_EV[1]  +8..11 W_EV[2]
//  +12..15 W_dh[0]  +16..19 W_dh[1]
//  +20,21 norm0{beta,gamma}  +22,23 norm1{beta,gamma}
// ---------------------------------------------------------------------------
extern "C" void kernel_launch(void* const* d_in, const int* in_sizes, int n_in,
                              void* d_out, int out_size, void* d_ws, size_t ws_size,
                              hipStream_t stream) {
  (void)in_sizes; (void)n_in; (void)out_size; (void)ws_size;
  const float* V     = (const float*)d_in[0];
  const float* E     = (const float*)d_in[1];
  const int*   E_idx = (const int*)d_in[2];
  const float* mask  = (const float*)d_in[3];

  // workspace carve (all offsets 32B-aligned)
  float*    hVa   = (float*)d_ws;
  float*    hVb   = hVa + (size_t)C_NODES * C_CH;
  _Float16* hEh   = (_Float16*)(hVb + (size_t)C_NODES * C_CH);
  _Float16* wpool = hEh + (size_t)C_EDGES * C_CH;

  // pack message-MLP weights into B-fragment tile order (f16, zero-padded)
  struct WInfo { int K, N; };
  static const WInfo WI[9] = {{48, 48}, {348, 100}, {48, 16},
                              {16, 16}, {116, 100}, {16, 16},
                              {16, 16}, {116, 100}, {16, 16}};
  LayerW lw[3];
  _Float16* wp = wpool;
  for (int l = 0; l < 3; ++l) {
    int base = 16 + l * 24;
    const int sidx[9] = {base + 0, base + 2, base + 3,
                         base + 4, base + 6, base + 7,
                         base + 8, base + 10, base + 11};
    _Float16* dsts[9];
    for (int k = 0; k < 9; ++k) {
      int K = WI[k].K, N = WI[k].N;
      int KT = (K + 31) / 32, NT = (N + 15) / 16;
      int total = KT * NT * 512;
      dsts[k] = wp;
      pack_b_kernel<<<(total + 255) / 256, 256, 0, stream>>>(
          (const float*)d_in[sidx[k]], wp, K, N, NT, total);
      wp += total;
    }
    lw[l].wh0 = dsts[0]; lw[l].ws0 = dsts[1]; lw[l].wv0 = dsts[2];
    lw[l].wh1 = dsts[3]; lw[l].ws1 = dsts[4]; lw[l].wv1 = dsts[5];
    lw[l].wh2 = dsts[6]; lw[l].ws2 = dsts[7]; lw[l].wv2 = dsts[8];
    lw[l].b0  = (const float*)d_in[base + 1];
    lw[l].b1  = (const float*)d_in[base + 5];
    lw[l].b2  = (const float*)d_in[base + 9];
    lw[l].be0 = (const float*)d_in[base + 20];
    lw[l].g0  = (const float*)d_in[base + 21];
  }

  // h_V = gvp(W_v, V)  (si=200)
  node_gvp_kernel<200><<<C_NODES / 256, 256, 0, stream>>>(
      V, (const float*)d_in[12], (const float*)d_in[13],
      (const float*)d_in[14], (const float*)d_in[15], hVa, C_NODES);
  // h_E = gvp(W_e, E) -> f16
  edge_gvp_kernel<<<C_EDGES / 256, 256, 0, stream>>>(
      E, (const float*)d_in[4], (const float*)d_in[5],
      (const float*)d_in[6], (const float*)d_in[7], hEh, C_EDGES);

  for (int l = 0; l < 3; ++l) {
    int base = 16 + l * 24;
    msg_layer_kernel<<<C_NODES, 128, SMEM_MSG, stream>>>(
        hVa, hEh, E_idx, mask, lw[l], hVb);
    ffn_kernel<<<C_NODES, 128, 0, stream>>>(
        hVb, mask,
        (const float*)d_in[base + 12], (const float*)d_in[base + 13],
        (const float*)d_in[base + 14], (const float*)d_in[base + 15],
        (const float*)d_in[base + 16], (const float*)d_in[base + 17],
        (const float*)d_in[base + 18], (const float*)d_in[base + 19],
        (const float*)d_in[base + 23], (const float*)d_in[base + 22],
        hVa);
  }

  // out = gvp(W_out, h_V)  (si=100)
  node_gvp_kernel<100><<<C_NODES / 256, 256, 0, stream>>>(
      hVa, (const float*)d_in[8], (const float*)d_in[9],
      (const float*)d_in[10], (const float*)d_in[11], (float*)d_out, C_NODES);
}
